// EncoderBlock_335007449314
// MI455X (gfx1250) — matrix operand order repaired and checked by reading, not compiled, not run on hardware
//
#include <hip/hip_runtime.h>
#include <hip/hip_bf16.h>
#include <math.h>

typedef __bf16 bf16;
typedef __attribute__((ext_vector_type(16))) __bf16 v16bf;
typedef __attribute__((ext_vector_type(8)))  __bf16 v8bf;
typedef __attribute__((ext_vector_type(8)))  float  v8f;

// Problem constants
#define Bq   4
#define Sq   2048
#define Dq   512
#define Hq   8
#define DKq  64
#define DFFq 2048
#define MSq  (Bq * Sq)   // 8192 rows

static __device__ inline v16bf cat16(v8bf lo, v8bf hi) {
  return __builtin_shufflevector(lo, hi, 0,1,2,3,4,5,6,7,8,9,10,11,12,13,14,15);
}

// CDNA5 async global->LDS copy (ASYNCcnt-tracked), per CDNA5 ISA
static __device__ inline void async_ld_b128(void* lds_ptr, const void* gptr) {
  unsigned l = (unsigned)(unsigned long long)lds_ptr;      // low 32b = LDS offset
  unsigned long long g = (unsigned long long)gptr;
  asm volatile("global_load_async_to_lds_b128 %0, %1, off" :: "v"(l), "v"(g) : "memory");
}
static __device__ inline void wait_async0() {
  asm volatile("s_wait_asynccnt 0" ::: "memory");
}

// ---------------------------------------------------------------------------
// f32 -> bf16 elementwise conversion (weights)
// ---------------------------------------------------------------------------
__global__ __launch_bounds__(256) void cvt_bf16_kernel(const float* __restrict__ in,
                                                       bf16* __restrict__ out, int n) {
  int i = blockIdx.x * 256 + threadIdx.x;
  if (i < n) out[i] = (bf16)in[i];
}

// ---------------------------------------------------------------------------
// LayerNorm (torch semantics: unbiased var, eps added to std), bf16 output.
// One row (512 f32) per 128-thread block.
// ---------------------------------------------------------------------------
__global__ __launch_bounds__(128) void ln_bf16_kernel(const float* __restrict__ x,
                                                      const float* __restrict__ g,
                                                      const float* __restrict__ be,
                                                      bf16* __restrict__ out) {
  __shared__ float red[128];
  const int row = blockIdx.x;
  const float* xr = x + (size_t)row * Dq;
  float v[4];
  float s = 0.f;
#pragma unroll
  for (int i = 0; i < 4; ++i) { v[i] = xr[threadIdx.x + i * 128]; s += v[i]; }
  red[threadIdx.x] = s; __syncthreads();
  for (int o = 64; o > 0; o >>= 1) {
    if (threadIdx.x < o) red[threadIdx.x] += red[threadIdx.x + o];
    __syncthreads();
  }
  const float mean = red[0] * (1.f / (float)Dq);
  __syncthreads();
  float ss = 0.f;
#pragma unroll
  for (int i = 0; i < 4; ++i) { float d = v[i] - mean; ss += d * d; }
  red[threadIdx.x] = ss; __syncthreads();
  for (int o = 64; o > 0; o >>= 1) {
    if (threadIdx.x < o) red[threadIdx.x] += red[threadIdx.x + o];
    __syncthreads();
  }
  const float var = red[0] * (1.f / (float)(Dq - 1));
  const float inv = 1.f / (sqrtf(var) + 1e-6f);
#pragma unroll
  for (int i = 0; i < 4; ++i) {
    int c = threadIdx.x + i * 128;
    out[(size_t)row * Dq + c] = (bf16)(g[c] * (v[i] - mean) * inv + be[c]);
  }
}

// ---------------------------------------------------------------------------
// WMMA bf16 GEMM: C[M,N] = A[M,K] * W[N,K]^T  (+resid) (relu) -> bf16/f32/QKV
// Block 256 thr (8 waves), tile 128x128, BK=32, double-buffered LDS filled by
// async global->LDS copies. Wave -> 32x64 (2x4 WMMA tiles), 8 WMMA / K-step.
// ---------------------------------------------------------------------------
template <int RELU, int RESID, int OUT_BF16, int QKV>
__global__ __launch_bounds__(256) void gemm_wmma_kernel(
    const bf16* __restrict__ A, const bf16* __restrict__ W,
    const float* __restrict__ resid, void* __restrict__ outp,
    int M, int N, int K) {
  __shared__ bf16 As[2][128][32];   // [buf][m][k]
  __shared__ bf16 Bs[2][128][32];   // [buf][n][k]

  const int tid  = threadIdx.x;
  const int lane = tid & 31;
  const int wave = tid >> 5;
  const int wm   = wave >> 1;     // 4 waves along M (32 rows each)
  const int wn   = wave & 1;      // 2 waves along N (64 cols each)
  const int half = lane >> 4;
  const int l16  = lane & 15;

  const int bm0 = blockIdx.y * 128;
  const int bn0 = blockIdx.x * 128;

  // async-stage one K-slab (128x32 A, 128x32 B) into LDS buffer `buf`
  auto stage = [&](int buf, int k0) {
#pragma unroll
    for (int it = 0; it < 2; ++it) {           // 512 v8-slots each for A and B
      int slot = tid + it * 256;
      int r = slot >> 2, c8 = slot & 3;
      async_ld_b128(&As[buf][r][c8 * 8],
                    &A[(size_t)(bm0 + r) * K + k0 + c8 * 8]);
      async_ld_b128(&Bs[buf][r][c8 * 8],
                    &W[(size_t)(bn0 + r) * K + k0 + c8 * 8]);
    }
  };

  v8f acc[2][4] = {};
  const int nk = K / 32;

  stage(0, 0);
  wait_async0();
  __syncthreads();

  for (int kt = 0; kt < nk; ++kt) {
    const int cur = kt & 1;
    if (kt + 1 < nk) stage(cur ^ 1, (kt + 1) * 32);   // overlap copy with WMMA

    v16bf af[2], bfr[4];
#pragma unroll
    for (int tm = 0; tm < 2; ++tm) {
      int m = wm * 32 + tm * 16 + l16;
      v8bf lo = *(const v8bf*)&As[cur][m][half * 16];
      v8bf hi = *(const v8bf*)&As[cur][m][half * 16 + 8];
      af[tm] = cat16(lo, hi);
    }
#pragma unroll
    for (int tn = 0; tn < 4; ++tn) {
      int n = wn * 64 + tn * 16 + l16;
      bfr[tn] = *(const v16bf*)&Bs[cur][n][half * 16];
    }
#pragma unroll
    for (int tm = 0; tm < 2; ++tm)
#pragma unroll
      for (int tn = 0; tn < 4; ++tn)
        acc[tm][tn] = __builtin_amdgcn_wmma_f32_16x16x32_bf16(
            false, af[tm], false, bfr[tn], (short)0, acc[tm][tn], false, false);

    wait_async0();      // prefetch of next slab has landed
    __syncthreads();    // everyone done reading `cur`, next iter may overwrite
  }

#pragma unroll
  for (int tm = 0; tm < 2; ++tm)
#pragma unroll
    for (int tn = 0; tn < 4; ++tn) {
      float* cp = (float*)&acc[tm][tn];
#pragma unroll
      for (int i = 0; i < 8; ++i) {
        int m = bm0 + wm * 32 + tm * 16 + half * 8 + i;
        int n = bn0 + wn * 64 + tn * 16 + l16;
        float v = cp[i];
        if (RELU)  v = v > 0.f ? v : 0.f;
        if (RESID) v += resid[(size_t)m * N + n];
        if (QKV) {  // scatter to [B,H,S,DK] bf16
          int b = m >> 11, s = m & (Sq - 1);
          int h = n >> 6,  dk = n & (DKq - 1);
          ((bf16*)outp)[(((size_t)(b * Hq + h)) * Sq + s) * DKq + dk] = (bf16)v;
        } else if (OUT_BF16) {
          ((bf16*)outp)[(size_t)m * N + n] = (bf16)v;
        } else {
          ((float*)outp)[(size_t)m * N + n] = v;
        }
      }
    }
}

// ---------------------------------------------------------------------------
// Flash attention: Q,K,V bf16 [B,H,S,DK] -> O bf16 [B,S,H,DK].
// Block 128 thr (4 waves); block owns 64 q rows; wave owns 16 q rows.
// Online softmax; K tile staged via async global->LDS.
// ---------------------------------------------------------------------------
__global__ __launch_bounds__(128) void fa_kernel(
    const bf16* __restrict__ Q, const bf16* __restrict__ Kp,
    const bf16* __restrict__ Vp, const int* __restrict__ mask,
    bf16* __restrict__ O) {
  __shared__ bf16  Ks[64][64];        // [k_seq][d]
  __shared__ bf16  Vt[64][72];        // [d][k_seq] (transposed, padded)
  __shared__ float Sbuf[4][16][64];   // per-wave score rows
  __shared__ bf16  Pbuf[4][16][64];   // per-wave probabilities (bf16)
  __shared__ float statsM[4][16], statsL[4][16], alphaB[4][16];
  __shared__ int   mk[64];

  const int tid  = threadIdx.x;
  const int lane = tid & 31;
  const int wave = tid >> 5;
  const int half = lane >> 4;
  const int l16  = lane & 15;

  const int qt = blockIdx.x;          // 64-row q tile
  const int h  = blockIdx.y;
  const int b  = blockIdx.z;
  const size_t bh = (size_t)(b * Hq + h) * Sq;

  // Q fragments for this wave's 16 rows (2 chunks of K-dim 32)
  const int qrow = qt * 64 + wave * 16 + l16;
  v16bf qa[2];
#pragma unroll
  for (int c = 0; c < 2; ++c) {
    const bf16* qp = Q + (bh + qrow) * DKq + c * 32 + half * 16;
    qa[c] = cat16(*(const v8bf*)qp, *(const v8bf*)(qp + 8));
  }

  v8f oacc[4] = {};
  if (lane < 16) { statsM[wave][lane] = -3.0e38f; statsL[wave][lane] = 0.f; }
  __syncthreads();

  for (int kt = 0; kt < Sq / 64; ++kt) {
    // Stage K tile via async copy; V tile transposed through VGPRs
#pragma unroll
    for (int it = 0; it < 4; ++it) {
      int slot = tid + it * 128;      // 512 v8-slots
      int r = slot >> 3, c8 = slot & 7;
      async_ld_b128(&Ks[r][c8 * 8], &Kp[(bh + kt * 64 + r) * DKq + c8 * 8]);
      v8bf vv = *(const v8bf*)&Vp[(bh + kt * 64 + r) * DKq + c8 * 8];
      const bf16* vpp = (const bf16*)&vv;
#pragma unroll
      for (int j = 0; j < 8; ++j) Vt[c8 * 8 + j][r] = vpp[j];
    }
    if (tid < 64) mk[tid] = mask[(size_t)b * Sq + kt * 64 + tid];
    wait_async0();
    __syncthreads();

    // S = Q K^T * 1/sqrt(DK), masked, to LDS
#pragma unroll
    for (int ns = 0; ns < 4; ++ns) {
      v8f s = {};
#pragma unroll
      for (int c = 0; c < 2; ++c) {
        v16bf kb = *(const v16bf*)&Ks[ns * 16 + l16][c * 32 + half * 16];
        s = __builtin_amdgcn_wmma_f32_16x16x32_bf16(false, qa[c], false, kb,
                                                    (short)0, s, false, false);
      }
      float* sp = (float*)&s;
      const int n = ns * 16 + l16;
#pragma unroll
      for (int i = 0; i < 8; ++i) {
        float v = sp[i] * 0.125f;
        if (mk[n] == 0) v = -3.0e38f;
        Sbuf[wave][half * 8 + i][n] = v;
      }
    }
    __syncthreads();

    // Online softmax, one lane per row
    if (lane < 16) {
      const int r = lane;
      float mOld = statsM[wave][r];
      float mNew = mOld;
#pragma unroll 8
      for (int j = 0; j < 64; ++j) mNew = fmaxf(mNew, Sbuf[wave][r][j]);
      float alpha = __expf(mOld - mNew);
      float sum = 0.f;
#pragma unroll 8
      for (int j = 0; j < 64; ++j) {
        float p = __expf(Sbuf[wave][r][j] - mNew);
        Pbuf[wave][r][j] = (bf16)p;
        sum += p;
      }
      statsM[wave][r] = mNew;
      statsL[wave][r] = statsL[wave][r] * alpha + sum;
      alphaB[wave][r] = alpha;
    }
    __syncthreads();

    // Rescale running O, then O += P @ V
#pragma unroll
    for (int t = 0; t < 4; ++t) {
      float* op = (float*)&oacc[t];
#pragma unroll
      for (int i = 0; i < 8; ++i) op[i] *= alphaB[wave][half * 8 + i];
    }
#pragma unroll
    for (int c = 0; c < 2; ++c) {
      const bf16* pp = &Pbuf[wave][l16][c * 32 + half * 16];
      v16bf pa = cat16(*(const v8bf*)pp, *(const v8bf*)(pp + 8));
#pragma unroll
      for (int t = 0; t < 4; ++t) {
        v16bf vb = *(const v16bf*)&Vt[t * 16 + l16][c * 32 + half * 16];
        oacc[t] = __builtin_amdgcn_wmma_f32_16x16x32_bf16(false, pa, false, vb,
                                                          (short)0, oacc[t],
                                                          false, false);
      }
    }
    __syncthreads();
  }

  // Finalize: divide by row sums, store [B,S,H,DK] bf16
#pragma unroll
  for (int t = 0; t < 4; ++t) {
    float* op = (float*)&oacc[t];
#pragma unroll
    for (int i = 0; i < 8; ++i) {
      int mrel = half * 8 + i;
      float v = op[i] / statsL[wave][mrel];
      int q = qt * 64 + wave * 16 + mrel;
      int d = t * 16 + l16;
      O[(((size_t)b * Sq + q) * Hq + h) * DKq + d] = (bf16)v;
    }
  }
}

// ---------------------------------------------------------------------------
// Orchestration
// ---------------------------------------------------------------------------
extern "C" void kernel_launch(void* const* d_in, const int* in_sizes, int n_in,
                              void* d_out, int out_size, void* d_ws, size_t ws_size,
                              hipStream_t stream) {
  (void)in_sizes; (void)n_in; (void)out_size; (void)ws_size;

  const float* x    = (const float*)d_in[0];
  const int*   mask = (const int*)d_in[1];
  const float* Wq   = (const float*)d_in[2];
  const float* Wk   = (const float*)d_in[3];
  const float* Wv   = (const float*)d_in[4];
  const float* Wo   = (const float*)d_in[5];
  const float* w1   = (const float*)d_in[6];
  const float* w2   = (const float*)d_in[7];
  const float* a1   = (const float*)d_in[8];
  const float* b1   = (const float*)d_in[9];
  const float* a2   = (const float*)d_in[10];
  const float* b2   = (const float*)d_in[11];

  char* ws = (char*)d_ws;
  size_t off = 0;
  auto take = [&](size_t bytes) -> char* {
    char* p = ws + off;
    off += (bytes + 255) & ~(size_t)255;
    return p;
  };

  bf16*  lnbuf = (bf16*)take((size_t)MSq * Dq * 2);   // LN1 out, reused for LN2 out
  bf16*  wqb   = (bf16*)take((size_t)Dq * Dq * 2);
  bf16*  wkb   = (bf16*)take((size_t)Dq * Dq * 2);
  bf16*  wvb   = (bf16*)take((size_t)Dq * Dq * 2);
  bf16*  wob   = (bf16*)take((size_t)Dq * Dq * 2);
  bf16*  w1b   = (bf16*)take((size_t)DFFq * Dq * 2);
  bf16*  w2b   = (bf16*)take((size_t)Dq * DFFq * 2);
  bf16*  Qb    = (bf16*)take((size_t)MSq * Dq * 2);
  bf16*  Kb    = (bf16*)take((size_t)MSq * Dq * 2);
  bf16*  Vb    = (bf16*)take((size_t)MSq * Dq * 2);
  bf16*  Ob    = (bf16*)take((size_t)MSq * Dq * 2);
  float* src   = (float*)take((size_t)MSq * Dq * 4);
  bf16*  f1    = Qb;  // FFN1 activations (8192x2048 bf16 = 32MB) reuse Q/K/V/O region

  // 1) weights -> bf16
  cvt_bf16_kernel<<<(Dq * Dq) / 256, 256, 0, stream>>>(Wq, wqb, Dq * Dq);
  cvt_bf16_kernel<<<(Dq * Dq) / 256, 256, 0, stream>>>(Wk, wkb, Dq * Dq);
  cvt_bf16_kernel<<<(Dq * Dq) / 256, 256, 0, stream>>>(Wv, wvb, Dq * Dq);
  cvt_bf16_kernel<<<(Dq * Dq) / 256, 256, 0, stream>>>(Wo, wob, Dq * Dq);
  cvt_bf16_kernel<<<(DFFq * Dq) / 256, 256, 0, stream>>>(w1, w1b, DFFq * Dq);
  cvt_bf16_kernel<<<(Dq * DFFq) / 256, 256, 0, stream>>>(w2, w2b, Dq * DFFq);

  // 2) LN1
  ln_bf16_kernel<<<MSq, 128, 0, stream>>>(x, a1, b1, lnbuf);

  // 3) Q/K/V projections, scatter to [B,H,S,DK]
  gemm_wmma_kernel<0, 0, 1, 1><<<dim3(Dq / 128, MSq / 128), 256, 0, stream>>>(
      lnbuf, wqb, nullptr, Qb, MSq, Dq, Dq);
  gemm_wmma_kernel<0, 0, 1, 1><<<dim3(Dq / 128, MSq / 128), 256, 0, stream>>>(
      lnbuf, wkb, nullptr, Kb, MSq, Dq, Dq);
  gemm_wmma_kernel<0, 0, 1, 1><<<dim3(Dq / 128, MSq / 128), 256, 0, stream>>>(
      lnbuf, wvb, nullptr, Vb, MSq, Dq, Dq);

  // 4) flash attention -> O [B,S,H,DK]
  fa_kernel<<<dim3(Sq / 64, Hq, Bq), 128, 0, stream>>>(Qb, Kb, Vb, mask, Ob);

  // 5) output projection + residual x -> src (f32)
  gemm_wmma_kernel<0, 1, 0, 0><<<dim3(Dq / 128, MSq / 128), 256, 0, stream>>>(
      Ob, wob, x, src, MSq, Dq, Dq);

  // 6) LN2
  ln_bf16_kernel<<<MSq, 128, 0, stream>>>(src, a2, b2, lnbuf);

  // 7) FFN1: relu(h @ w1^T) -> bf16
  gemm_wmma_kernel<1, 0, 1, 0><<<dim3(DFFq / 128, MSq / 128), 256, 0, stream>>>(
      lnbuf, w1b, nullptr, f1, MSq, DFFq, Dq);

  // 8) FFN2 + residual src -> d_out (f32)
  gemm_wmma_kernel<0, 1, 0, 0><<<dim3(Dq / 128, MSq / 128), 256, 0, stream>>>(
      f1, w2b, src, (float*)d_out, MSq, Dq, DFFq);
}
